// RNN_21663815041520
// MI455X (gfx1250) — compile-verified, matrix operand order checked by reading, not run on hardware
//
#include <hip/hip_runtime.h>
#include <cstdint>
#include <cstddef>

// ---------------- problem constants ----------------
constexpr int B_SZ  = 8;
constexpr int S_LEN = 4096;
constexpr int D_IN  = 2048;
constexpr int D_OUT = 2048;
constexpr int NHEAD = 32;   // N
constexpr int HDIM  = 64;   // H
constexpr int E_DIM = NHEAD * HDIM; // 2048

// ---------------- vector types ----------------
typedef __attribute__((ext_vector_type(16))) __bf16 v16bf;
typedef __attribute__((ext_vector_type(8)))  float  v8f;
typedef __attribute__((ext_vector_type(2)))  __bf16 bf16x2;
typedef __attribute__((ext_vector_type(8)))  __bf16 bf16x8;
typedef __attribute__((ext_vector_type(4)))  float  f32x4;
typedef __attribute__((ext_vector_type(4)))  unsigned int u32x4;
typedef __attribute__((ext_vector_type(4)))  int    i32x4;
typedef __attribute__((ext_vector_type(8)))  int    i32x8;

union Frag16 { v16bf v; bf16x2 p[8]; };

// K offset inside a 16x32 bf16 A/B fragment for VGPR v, lane-half 'half'
// (ISA 7.12.2: lanes0-15 v0..3 -> K{0,2,4,6}, v4..7 -> K{16,18,20,22}; lanes16-31 +8)
__device__ __forceinline__ int frag_k(int v, int half) {
    return ((v & 3) * 2) + ((v >> 2) * 16) + half * 8;
}

__device__ __forceinline__ v8f wmma_bf16(const Frag16& a, const Frag16& b, v8f c) {
    return __builtin_amdgcn_wmma_f32_16x16x32_bf16(
        false, a.v, false, b.v, (short)0, c, false, false);
}

// Branchless fast tanh: 1 - 2/(exp2(2*log2e*x)+1). Saturates correctly at +-1,
// ~4 VALU ops (v_exp_f32 + v_rcp_f32), no EXEC-mask branches on the scan's
// sequential critical path.
__device__ __forceinline__ float fast_tanh(float x) {
    float e = __builtin_amdgcn_exp2f(x * 2.885390081777927f); // 2/ln(2)
    return 1.0f - 2.0f * __builtin_amdgcn_rcpf(e + 1.0f);
}

// ----------------------------------------------------------------------
// Tensor Data Mover: 2D tile load (bf16) global -> LDS, per ISA 08 §8.
// Describes exactly the tile (tensor dims == tile dims), row stride in
// elements, LDS padding: 16 DWORDs data (64B row) + 4 DWORDs pad -> 80B rows.
// ----------------------------------------------------------------------
__device__ __forceinline__ void tdm_load_tile_bf16(uint64_t gaddr, unsigned lds_off,
                                                   unsigned tile_k, unsigned tile_m,
                                                   unsigned row_stride_elems)
{
    u32x4 g0;
    g0[0] = 1u;                                               // count=1 (valid user D#)
    g0[1] = lds_off;                                          // lds_addr (bytes)
    g0[2] = (unsigned)(gaddr & 0xFFFFFFFFu);                  // global_addr[31:0]
    g0[3] = (unsigned)((gaddr >> 32) & 0x1FFFFFFu) | (2u << 30); // addr[56:32] | type=2
    i32x8 g1;
    g1[0] = (int)((1u << 16)            // data_size = 1 -> 2 bytes
                | (1u << 20)            // pad_enable
                | (3u << 22)            // pad_interval: code 3 -> 16 DWORDs (64B)
                | (3u << 25));          // pad_amount:   code 3 -> 4 DWORDs (16B)
    g1[1] = (int)((tile_k & 0xFFFFu) << 16);                  // tensor_dim0[15:0] (bits 63:48)
    g1[2] = (int)((tile_k >> 16) | ((tile_m & 0xFFFFu) << 16)); // dim0 hi | tensor_dim1 lo
    g1[3] = (int)(((tile_m >> 16) & 0xFFFFu) | ((tile_k & 0xFFFFu) << 16)); // dim1 hi | tile_dim0
    g1[4] = (int)(tile_m & 0xFFFFu);                          // tile_dim1 | tile_dim2=0
    g1[5] = (int)row_stride_elems;                            // tensor_dim0_stride[31:0]
    g1[6] = 0;                                                // stride hi | dim1_stride lo
    g1[7] = 0;
    const i32x4 z4 = {0, 0, 0, 0};
#if __clang_major__ >= 23
    const i32x8 z8 = {0, 0, 0, 0, 0, 0, 0, 0};
    __builtin_amdgcn_tensor_load_to_lds(g0, g1, z4, z4, z8, 0);
#else
    __builtin_amdgcn_tensor_load_to_lds(g0, g1, z4, z4, 0);
#endif
}

// ======================================================================
// GEMM + bias:  C[M,Nout] = A[M,K] * W[Nout,K]^T + bias
// Block 256 threads = 8 waves. Tile BM=64, BN=128, BK=32, wave: 2x2 WMMAs.
// Double-buffered LDS, register-staged prefetch, 1 barrier/iteration.
// USE_TDM: A tiles (bf16 source) moved by the Tensor Data Mover.
// ======================================================================
template <bool A_BF16, bool OUT_BF16, bool USE_TDM>
__global__ __launch_bounds__(256, 2)
void gemm_bias_wmma(const void* __restrict__ Asrc,
                    const float* __restrict__ Wsrc,   // (Nout, K) row-major
                    const float* __restrict__ bias,   // (Nout)
                    void* __restrict__ Cdst,
                    int M, int K, int Nout)
{
    constexpr int BM = 64, BN = 128, BK = 32, LDT = BK + 8; // 80B rows, 16B aligned
    __shared__ __bf16 As[2][BM][LDT];
    __shared__ __bf16 Bs[2][BN][LDT];   // N-major: Bs[buf][n][k]

    const int tid  = threadIdx.x;
    const int lane = tid & 31;
    const int wid  = tid >> 5;   // 0..7
    const int wm   = wid >> 2;   // 0..1
    const int wn   = wid & 3;    // 0..3
    const int lm   = lane & 15;
    const int half = lane >> 4;

    const int n0 = blockIdx.x * BN;
    const int m0 = blockIdx.y * BM;

    int kv[8];
    #pragma unroll
    for (int v = 0; v < 8; ++v) kv[v] = frag_k(v, half);

    const v8f zero8 = {0.f,0.f,0.f,0.f,0.f,0.f,0.f,0.f};
    v8f c[2][2];
    #pragma unroll
    for (int i = 0; i < 2; ++i)
        #pragma unroll
        for (int j = 0; j < 2; ++j) c[i][j] = zero8;

    const int arow = tid >> 2, akc = (tid & 3) * 8;   // A: 64 rows x (4 thr * 8 elems)
    const int brow = tid >> 1, bkc = (tid & 1) * 16;  // B: 128 rows x (2 thr * 16 elems)

    bf16x8 bst0, bst1, ast;   // register staging for the next tile

    auto load_regs = [&](int k0) {
        const size_t bbase = (size_t)(n0 + brow) * K + k0 + bkc;
        f32x4 w0 = *(const f32x4*)(Wsrc + bbase);
        f32x4 w1 = *(const f32x4*)(Wsrc + bbase + 4);
        f32x4 w2 = *(const f32x4*)(Wsrc + bbase + 8);
        f32x4 w3 = *(const f32x4*)(Wsrc + bbase + 12);
        #pragma unroll
        for (int i = 0; i < 4; ++i) {
            bst0[i] = (__bf16)w0[i]; bst0[4 + i] = (__bf16)w1[i];
            bst1[i] = (__bf16)w2[i]; bst1[4 + i] = (__bf16)w3[i];
        }
        if constexpr (!USE_TDM) {
            const size_t abase = (size_t)(m0 + arow) * K + k0 + akc;
            if constexpr (A_BF16) {
                ast = *(const bf16x8*)((const __bf16*)Asrc + abase);
            } else {
                const float* Af = (const float*)Asrc;
                f32x4 a0 = *(const f32x4*)(Af + abase);
                f32x4 a1 = *(const f32x4*)(Af + abase + 4);
                #pragma unroll
                for (int i = 0; i < 4; ++i) { ast[i] = (__bf16)a0[i]; ast[4 + i] = (__bf16)a1[i]; }
            }
        }
    };
    auto store_lds = [&](int buf) {
        *(bf16x8*)&Bs[buf][brow][bkc]     = bst0;
        *(bf16x8*)&Bs[buf][brow][bkc + 8] = bst1;
        if constexpr (!USE_TDM) *(bf16x8*)&As[buf][arow][akc] = ast;
    };
    auto tdm_issue = [&](int k0, int buf) {
        if (wid == 0) {
            const uint64_t ga = (uint64_t)(uintptr_t)Asrc +
                                ((size_t)m0 * K + k0) * sizeof(__bf16);
            // generic shared pointer -> LDS byte offset = addr[31:0] (ISA 10.2)
            const unsigned loff = (unsigned)(uintptr_t)&As[buf][0][0];
            tdm_load_tile_bf16(ga, loff, BK, BM, (unsigned)K);
        }
    };

    const int niter = K / BK;
    // ---- prologue: tile 0 ----
    load_regs(0);
    store_lds(0);
    if constexpr (USE_TDM) tdm_issue(0, 0);

    for (int i = 0; i < niter; ++i) {
        const int cur = i & 1;
        if constexpr (USE_TDM) {
            if (wid == 0) __builtin_amdgcn_s_wait_tensorcnt(0);  // TDM(tile i) done
        }
        __syncthreads();   // buf[cur] visible to all waves; prior reads of buf[1-cur] done

        if (i + 1 < niter) {
            load_regs((i + 1) * BK);                 // global loads in flight over compute
            if constexpr (USE_TDM) tdm_issue((i + 1) * BK, 1 - cur);
        }

        // ---- gather fragments from buf[cur], 4 WMMAs ----
        Frag16 af[2], bfg[2];
        #pragma unroll
        for (int mt = 0; mt < 2; ++mt) {
            const __bf16* rp = &As[cur][wm * 32 + mt * 16 + lm][0];
            #pragma unroll
            for (int v = 0; v < 8; ++v) af[mt].p[v] = *(const bf16x2*)(rp + kv[v]);
        }
        #pragma unroll
        for (int nt = 0; nt < 2; ++nt) {
            const __bf16* rp = &Bs[cur][wn * 32 + nt * 16 + lm][0];
            #pragma unroll
            for (int v = 0; v < 8; ++v) bfg[nt].p[v] = *(const bf16x2*)(rp + kv[v]);
        }
        #pragma unroll
        for (int mt = 0; mt < 2; ++mt)
            #pragma unroll
            for (int nt = 0; nt < 2; ++nt)
                c[mt][nt] = wmma_bf16(af[mt], bfg[nt], c[mt][nt]);

        if (i + 1 < niter) store_lds(1 - cur);       // stage next tile (other buffer)
    }

    // ---- epilogue: + bias, store (C/D layout: lane n=l&15, VGPR v -> m=v+half*8) ----
    #pragma unroll
    for (int nt = 0; nt < 2; ++nt) {
        const int ncol = n0 + wn * 32 + nt * 16 + lm;
        const float bv = bias[ncol];
        #pragma unroll
        for (int mt = 0; mt < 2; ++mt) {
            #pragma unroll
            for (int v = 0; v < 8; ++v) {
                const int mrow = m0 + wm * 32 + mt * 16 + half * 8 + v;
                const float val = c[mt][nt][v] + bv;
                if constexpr (OUT_BF16)
                    ((__bf16*)Cdst)[(size_t)mrow * Nout + ncol] = (__bf16)val;
                else
                    ((float*)Cdst)[(size_t)mrow * Nout + ncol] = val;
            }
        }
    }
}

// ======================================================================
// Recurrent scan: h = tanh(h @ Wg + x_t), per weight-group g.
// Block = (group g, batch-half bh): 16 rows = 4 batches x 4 heads (shared Wg).
// 128 threads = 4 waves; wave w owns output columns [16w,16w+16).
// W fragments live in registers; h double-buffered in LDS -> 1 barrier/step;
// chain of only 2 WMMAs per step; branchless tanh; x_{t+1} prefetched.
// ======================================================================
__global__ __launch_bounds__(128, 1)
void rnn_scan_wmma(__bf16* y,                        // (B*S, 2048) bf16: xp in, y out
                   const float* __restrict__ Wsrc,   // (8, 64, 64): [g][h_in][k_out]
                   float* __restrict__ hT)           // (B, 2048) f32
{
    const int g    = blockIdx.x >> 1;
    const int bh   = blockIdx.x & 1;
    const int tid  = threadIdx.x;
    const int lane = tid & 31;
    const int wid  = tid >> 5;          // 0..3
    const int lm   = lane & 15;
    const int half = lane >> 4;
    const int ncol = wid * 16 + lm;     // 0..63

    __shared__ __bf16 hs[2][16][64];    // double-buffered h state
    for (int i = tid; i < 16 * 64; i += 128) (&hs[0][0][0])[i] = (__bf16)0.f;

    // Preload invariant B fragments of Wg (K=64 -> 2 fragments of 32)
    const float* Wg = Wsrc + (size_t)g * HDIM * HDIM;
    Frag16 bw[2];
    #pragma unroll
    for (int ks = 0; ks < 2; ++ks) {
        #pragma unroll
        for (int v = 0; v < 8; ++v) {
            const int k = ks * 32 + frag_k(v, half);
            bf16x2 pr;
            pr[0] = (__bf16)Wg[(size_t)k * HDIM + ncol];
            pr[1] = (__bf16)Wg[(size_t)(k + 1) * HDIM + ncol];
            bw[ks].p[v] = pr;
        }
    }

    // Per-accumulator global element index for x_t / y (D tile: m = half*8 + v)
    size_t xbase[8];
    #pragma unroll
    for (int v = 0; v < 8; ++v) {
        const int m  = half * 8 + v;
        const int b  = bh * 4 + (m >> 2);
        const int nh = g * 4 + (m & 3);
        xbase[v] = (size_t)b * S_LEN * E_DIM + (size_t)nh * HDIM + ncol;
    }

    __bf16 xc[8];
    #pragma unroll
    for (int v = 0; v < 8; ++v) xc[v] = y[xbase[v]];   // prefetch t=0

    __syncthreads();   // hs[0] zero visible

    const v8f zero8 = {0.f,0.f,0.f,0.f,0.f,0.f,0.f,0.f};
    float dv[8] = {0.f,0.f,0.f,0.f,0.f,0.f,0.f,0.f};
    int cur = 0;

    for (int t = 0; t < S_LEN; ++t) {
        // gather A fragments (h) from LDS buffer 'cur'
        Frag16 ah[2];
        #pragma unroll
        for (int ks = 0; ks < 2; ++ks) {
            const __bf16* rp = &hs[cur][lm][ks * 32];
            #pragma unroll
            for (int v = 0; v < 8; ++v)
                ah[ks].p[v] = *(const bf16x2*)(rp + frag_k(v, half));
        }

        v8f acc = zero8;
        acc = wmma_bf16(ah[0], bw[0], acc);
        acc = wmma_bf16(ah[1], bw[1], acc);

        // prefetch x_{t+1} while WMMAs are in flight
        __bf16 xn[8];
        if (t + 1 < S_LEN) {
            #pragma unroll
            for (int v = 0; v < 8; ++v) xn[v] = y[xbase[v] + (size_t)(t + 1) * E_DIM];
        } else {
            #pragma unroll
            for (int v = 0; v < 8; ++v) xn[v] = (__bf16)0.f;
        }

        #pragma unroll
        for (int v = 0; v < 8; ++v) dv[v] = fast_tanh(acc[v] + (float)xc[v]);

        // write y_t (overwrite consumed xp slot, bf16 for the output GEMM)
        #pragma unroll
        for (int v = 0; v < 8; ++v) y[xbase[v] + (size_t)t * E_DIM] = (__bf16)dv[v];

        // publish h_t into the other buffer; single barrier per step
        #pragma unroll
        for (int v = 0; v < 8; ++v) hs[1 - cur][half * 8 + v][ncol] = (__bf16)dv[v];
        __syncthreads();
        cur ^= 1;

        #pragma unroll
        for (int v = 0; v < 8; ++v) xc[v] = xn[v];
    }

    // final hidden state -> hT (f32)
    #pragma unroll
    for (int v = 0; v < 8; ++v) {
        const int m  = half * 8 + v;
        const int b  = bh * 4 + (m >> 2);
        const int nh = g * 4 + (m & 3);
        hT[(size_t)b * E_DIM + (size_t)nh * HDIM + ncol] = dv[v];
    }
}

// ======================================================================
extern "C" void kernel_launch(void* const* d_in, const int* in_sizes, int n_in,
                              void* d_out, int out_size, void* d_ws, size_t ws_size,
                              hipStream_t stream)
{
    const float* x  = (const float*)d_in[0];   // (B,S,D_IN)
    const float* Wi = (const float*)d_in[1];   // (E, D_IN)
    const float* bi = (const float*)d_in[2];   // (E)
    const float* Ws = (const float*)d_in[3];   // (8,64,64)
    const float* Wo = (const float*)d_in[4];   // (D_OUT, E)
    const float* bo = (const float*)d_in[5];   // (D_OUT)

    float*  out = (float*)d_out;                               // (B,S,D_OUT)
    float*  hT  = out + (size_t)B_SZ * S_LEN * D_OUT;          // (B, E)
    __bf16* yb  = (__bf16*)d_ws;                               // (B*S, E) bf16

    const int M = B_SZ * S_LEN;   // 32768
    dim3 blk(256);
    dim3 grd(E_DIM / 128, M / 64);   // (16, 512)

    // 1) input projection: xp = x @ Wi^T + bi  -> bf16 ws (f32 source, VALU convert)
    gemm_bias_wmma<false, true, false><<<grd, blk, 0, stream>>>(x, Wi, bi, yb, M, D_IN, E_DIM);

    // 2) sequential RNN scan (in-place on ws), emits hT
    rnn_scan_wmma<<<dim3(16), dim3(128), 0, stream>>>(yb, Ws, hT);

    // 3) output projection: out = y @ Wo^T + bo -> f32 d_out (bf16 A tiles via TDM)
    gemm_bias_wmma<true, false, true><<<grd, blk, 0, stream>>>(yb, Wo, bo, out, M, E_DIM, D_OUT);
}
